// LocalTransformer_54133767799361
// MI455X (gfx1250) — matrix operand order repaired and checked by reading, not compiled, not run on hardware
//
#include <hip/hip_runtime.h>
#include <hip/hip_bf16.h>

#define BB   4
#define NN   8192
#define NPT  2048
#define NSAM 32
#define CC   256
#define DOUT 256
#define HH   4
#define LL   2
#define DH   64
#define R2F  0.16f
#define EPSF 1e-5f

typedef __attribute__((ext_vector_type(16))) __bf16 v16bf;
typedef __attribute__((ext_vector_type(8)))  __bf16 v8bf;
typedef __attribute__((ext_vector_type(8)))  float  v8f;

// ---------- helpers ----------
__device__ __forceinline__ __bf16 f2bf(float f) {
    union { unsigned short u; __bf16 b; } cv;
    unsigned u32 = __float_as_uint(f);
    cv.u = (unsigned short)((u32 + 0x7FFFu + ((u32 >> 16) & 1u)) >> 16);
    return cv.b;
}

// A operand: 16x32 bf16 tile, row-major source (LDS):
// lane L<16: row M=L, K = 0..15; lane L+16: K = 16..31 (matches load_Bt).
__device__ __forceinline__ v16bf load_A_lds(const __bf16* base, int ld, int lane) {
    int half = lane >> 4, m = lane & 15;
    const __bf16* row = base + m * ld + half * 16;
    return *(const v16bf*)(row);
}

// B operand: 32x16 tile of W (row-major [n][k]); lane L<16: col N=L, K=0..15;
// lane L+16: col N=L, K=16..31. Contiguous 32B per lane.
__device__ __forceinline__ v16bf load_Bt(const __bf16* W, int ldk, int n0, int k0, int lane) {
    int n  = n0 + (lane & 15);
    int kb = k0 + ((lane >> 4) << 4);
    return *(const v16bf*)(W + n * ldk + kb);
}

__device__ __forceinline__ v8f wmma_bf(v16bf a, v16bf b, v8f c) {
    return __builtin_amdgcn_wmma_f32_16x16x32_bf16(false, a, false, b, (short)0, c, false, false);
}

#define GMODE_ADD_F32 0
#define GMODE_RELU_BF 1

// out[m,n] = (HASC? Csrc[m,n]:0) + bias[n] + sum_k A[m,k]*W[n,k]
// M=32, N=NTILES*16, K=KT*32. All structural params compile-time -> branchless codegen.
template<int KT, int NTILES, bool HASC, int MODE>
__device__ __forceinline__ void gemm_rows(
    const __bf16* sA, int lda, const __bf16* W, int ldk, const float* bias,
    const float* Csrc, int ldc, float* outF, __bf16* outB, int ldo,
    int wave, int lane)
{
    constexpr int TOTAL = 2 * NTILES;
#pragma unroll 1
    for (int t = wave; t < TOTAL; t += 8) {
        int mt = t & 1, nt = t >> 1;
        int n = nt * 16 + (lane & 15);
        int mbase = mt * 16 + ((lane >> 4) << 3);
        float bi = bias[n];
        v8f acc;
        if (HASC) {
            // 8 unconditional LDS loads -> single wait, no scalar branches
            float cin[8];
#pragma unroll
            for (int r = 0; r < 8; ++r) cin[r] = Csrc[(mbase + r) * ldc + n];
#pragma unroll
            for (int r = 0; r < 8; ++r) acc[r] = bi + cin[r];
        } else {
#pragma unroll
            for (int r = 0; r < 8; ++r) acc[r] = bi;
        }
#pragma unroll
        for (int kt = 0; kt < KT; ++kt) {
            v16bf a = load_A_lds(sA + mt * 16 * lda + kt * 32, lda, lane);
            v16bf b = load_Bt(W, ldk, nt * 16, kt * 32, lane);
            acc = wmma_bf(a, b, acc);
        }
#pragma unroll
        for (int r = 0; r < 8; ++r) {
            int m = mbase + r;
            if (MODE == GMODE_ADD_F32) outF[m * ldo + n] = acc[r];
            else                       outB[m * ldo + n] = f2bf(fmaxf(acc[r], 0.0f));
        }
    }
}

// ---------- weight conversion ----------
__global__ __launch_bounds__(256) void cvt_bf16_kernel(const float* __restrict__ s,
                                                       __bf16* __restrict__ d, int n) {
    int i = blockIdx.x * 256 + threadIdx.x;
    if (i < n) d[i] = f2bf(s[i]);
}

// ---------- FPS (one block per batch, sequential 2048-step argmax) ----------
__global__ __launch_bounds__(1024) void fps_kernel(const float* __restrict__ xyz,
                                                   float* __restrict__ nxyz_ws,
                                                   float* __restrict__ out_newxyz,
                                                   float* __restrict__ out_fpsidx) {
    int b = blockIdx.x, tid = threadIdx.x;
    const float* xb = xyz + (size_t)b * NN * 3;
    float px[8], py[8], pz[8], dist[8];
#pragma unroll
    for (int i = 0; i < 8; ++i) {
        int p = tid + i * 1024;
        px[i] = xb[p * 3 + 0]; py[i] = xb[p * 3 + 1]; pz[i] = xb[p * 3 + 2];
        dist[i] = 1e10f;
    }
    __shared__ float s_far[3];
    __shared__ int   s_next;
    __shared__ float s_wmax[32];
    __shared__ int   s_widx[32];
    if (tid == 0) s_next = 0;
    __syncthreads();
    for (int it = 0; it < NPT; ++it) {
        int far = s_next;
        if (tid == 0) {
            float fx = xb[far * 3 + 0], fy = xb[far * 3 + 1], fz = xb[far * 3 + 2];
            s_far[0] = fx; s_far[1] = fy; s_far[2] = fz;
            int o = b * NPT + it;
            nxyz_ws[o * 3 + 0] = fx; nxyz_ws[o * 3 + 1] = fy; nxyz_ws[o * 3 + 2] = fz;
            out_newxyz[o * 3 + 0] = fx; out_newxyz[o * 3 + 1] = fy; out_newxyz[o * 3 + 2] = fz;
            out_fpsidx[o] = (float)far;
        }
        __syncthreads();
        float fx = s_far[0], fy = s_far[1], fz = s_far[2];
        float bmax = -1.0f; int bidx = 0x7fffffff;
#pragma unroll
        for (int i = 0; i < 8; ++i) {
            float dx = px[i] - fx, dy = py[i] - fy, dz = pz[i] - fz;
            float d = dx * dx + dy * dy + dz * dz;
            dist[i] = fminf(dist[i], d);
            int p = tid + i * 1024;
            if (dist[i] > bmax || (dist[i] == bmax && p < bidx)) { bmax = dist[i]; bidx = p; }
        }
#pragma unroll
        for (int off = 16; off >= 1; off >>= 1) {
            float om = __shfl_xor(bmax, off, 32);
            int   oi = __shfl_xor(bidx, off, 32);
            if (om > bmax || (om == bmax && oi < bidx)) { bmax = om; bidx = oi; }
        }
        int wv = tid >> 5;
        if ((tid & 31) == 0) { s_wmax[wv] = bmax; s_widx[wv] = bidx; }
        __syncthreads();
        if (wv == 0) {
            bmax = s_wmax[tid & 31]; bidx = s_widx[tid & 31];
#pragma unroll
            for (int off = 16; off >= 1; off >>= 1) {
                float om = __shfl_xor(bmax, off, 32);
                int   oi = __shfl_xor(bidx, off, 32);
                if (om > bmax || (om == bmax && oi < bidx)) { bmax = om; bidx = oi; }
            }
            if (tid == 0) s_next = bidx;
        }
        __syncthreads();
    }
}

// ---------- ball query: one wave32 per center ----------
__global__ __launch_bounds__(256) void ballq_kernel(const float* __restrict__ xyz,
                                                    const float* __restrict__ nxyz,
                                                    int* __restrict__ idx) {
    int gid  = blockIdx.x * 8 + (threadIdx.x >> 5);
    int lane = threadIdx.x & 31;
    int b = gid / NPT;
    const float* xb = xyz + (size_t)b * NN * 3;
    float cx = nxyz[gid * 3 + 0], cy = nxyz[gid * 3 + 1], cz = nxyz[gid * 3 + 2];
    int* out = idx + (size_t)gid * NSAM;
    int cnt = 0, first = 0; bool have = false;
    for (int j0 = 0; j0 < NN && cnt < NSAM; j0 += 32) {
        int j = j0 + lane;
        float dx = xb[j * 3 + 0] - cx, dy = xb[j * 3 + 1] - cy, dz = xb[j * 3 + 2] - cz;
        bool hit = (dx * dx + dy * dy + dz * dz) < R2F;
        unsigned mask = (unsigned)__ballot(hit);
        if (hit) {
            int pos = cnt + __builtin_popcount(mask & ((1u << lane) - 1u));
            if (pos < NSAM) out[pos] = j;
        }
        if (!have && mask) { first = j0 + __builtin_ctz(mask); have = true; }
        cnt += __builtin_popcount(mask);
    }
    if (cnt > NSAM) cnt = NSAM;
    if (cnt == 0) { if (lane == 0) out[0] = 0; cnt = 1; first = 0; }
    for (int p = cnt + lane; p < NSAM; p += 32) out[p] = first;
}

// ---------- layernorm over C=256, 8 threads/token ----------
__device__ __forceinline__ void layernorm_ws(float* s_x, __bf16* s_a,
                                             const float* g, const float* bvec, int tid) {
    int tok = tid >> 3, sub = tid & 7;
    float s1 = 0.f, s2 = 0.f;
#pragma unroll
    for (int i = 0; i < 32; ++i) {
        float v = s_x[tok * CC + sub * 32 + i];
        s1 += v; s2 += v * v;
    }
#pragma unroll
    for (int off = 1; off < 8; off <<= 1) {
        s1 += __shfl_xor(s1, off, 32);
        s2 += __shfl_xor(s2, off, 32);
    }
    float mean = s1 * (1.0f / CC);
    float var  = s2 * (1.0f / CC) - mean * mean;
    float rs   = rsqrtf(var + EPSF);
#pragma unroll
    for (int i = 0; i < 32; ++i) {
        int c = sub * 32 + i;
        float v = (s_x[tok * CC + c] - mean) * rs * g[c] + bvec[c];
        s_x[tok * CC + c] = v;
        s_a[tok * CC + c] = f2bf(v);
    }
}

// ---------- main per-group kernel ----------
__global__ __launch_bounds__(256) void group_kernel(
    const float* __restrict__ features, const float* __restrict__ xyz,
    const float* __restrict__ nxyz, const int* __restrict__ idx,
    const float* __restrict__ pe1_w, const float* __restrict__ pe1_b,
    const float* __restrict__ bn_g, const float* __restrict__ bn_b,
    const float* __restrict__ bn_m, const float* __restrict__ bn_v,
    const __bf16* __restrict__ pe2_wb, const float* __restrict__ pe2_b,
    const float* __restrict__ ln1_g, const float* __restrict__ ln1_b,
    const __bf16* __restrict__ qkv_wb, const float* __restrict__ qkv_b,
    const __bf16* __restrict__ out_wb, const float* __restrict__ out_b,
    const float* __restrict__ ln2_g, const float* __restrict__ ln2_b,
    const __bf16* __restrict__ ff1_wb, const float* __restrict__ ff1_b,
    const __bf16* __restrict__ ff2_wb, const float* __restrict__ ff2_b,
    const float* __restrict__ fc_w, const float* __restrict__ fc_b,
    float* __restrict__ out)
{
    __shared__ __align__(32) float  s_x[NSAM * CC];          // 32KB fp32 activations
    __shared__ __align__(32) __bf16 s_a[NSAM * CC];          // 16KB bf16 A operand / O
    __shared__ __align__(32) __bf16 s_qkv[NSAM * 3 * CC];    // 48KB q|k|v^T / pe-h / ff-hidden
    __shared__ __align__(32) __bf16 s_p[HH * NSAM * NSAM];   // 8KB attn probs
    __shared__ int   s_idx[NSAM];
    __shared__ float s_gx[NSAM * 3];
    __shared__ float s_pool[CC];

    int gid = blockIdx.x;
    int b = gid / NPT, g = gid % NPT;
    int tid = threadIdx.x, lane = tid & 31, wave = tid >> 5;

    // warm L2 paths for the first weight tiles (global_prefetch_b8)
    __builtin_prefetch(qkv_wb + (size_t)tid * 64, 0, 1);
    __builtin_prefetch(ff1_wb + (size_t)tid * 64, 0, 1);

    if (tid < NSAM) s_idx[tid] = idx[(size_t)gid * NSAM + tid];
    __syncthreads();

    if (tid < NSAM) {
        int p = s_idx[tid];
        s_gx[tid * 3 + 0] = xyz[((size_t)b * NN + p) * 3 + 0] - nxyz[gid * 3 + 0];
        s_gx[tid * 3 + 1] = xyz[((size_t)b * NN + p) * 3 + 1] - nxyz[gid * 3 + 1];
        s_gx[tid * 3 + 2] = xyz[((size_t)b * NN + p) * 3 + 2] - nxyz[gid * 3 + 2];
    }
    const float* fb = features + (size_t)b * CC * NN;
    for (int t = tid; t < NSAM * CC; t += 256) {
        int tok = t >> 8, c = t & 255;
        s_x[t] = fb[(size_t)c * NN + s_idx[tok]];      // g_feat gather (L2-resident)
    }
    __syncthreads();

    // PE1 -> BN -> ReLU -> bf16 h [32][128] staged in s_qkv
    for (int t = tid; t < NSAM * 128; t += 256) {
        int tok = t >> 7, j = t & 127;
        float hv = s_gx[tok * 3 + 0] * pe1_w[j * 3 + 0] +
                   s_gx[tok * 3 + 1] * pe1_w[j * 3 + 1] +
                   s_gx[tok * 3 + 2] * pe1_w[j * 3 + 2] + pe1_b[j];
        hv = (hv - bn_m[j]) * rsqrtf(bn_v[j] + EPSF) * bn_g[j] + bn_b[j];
        s_qkv[t] = f2bf(fmaxf(hv, 0.0f));
    }
    __syncthreads();
    // x = g_feat + h @ pe2^T + pe2_b   (WMMA, K=128)
    gemm_rows<4, 16, true, GMODE_ADD_F32>(s_qkv, 128, pe2_wb, 128, pe2_b,
                                          s_x, CC, s_x, nullptr, CC, wave, lane);
    __syncthreads();

    const float scale = 0.125f; // 1/sqrt(64)
    for (int l = 0; l < LL; ++l) {
        layernorm_ws(s_x, s_a, ln1_g + l * CC, ln1_b + l * CC, tid);
        __syncthreads();

        // ---- QKV: [32,768] = s_a @ qkv_w^T + b;  q rows | k rows | v transposed ----
        const __bf16* Wq = qkv_wb + (size_t)l * 768 * CC;
#pragma unroll 1
        for (int t = wave; t < 96; t += 8) {
            int mt = t & 1, nt = t >> 1;
            int nl = lane & 15;
            int n = nt * 16 + nl;
            int mbase = mt * 16 + ((lane >> 4) << 3);
            float bi = qkv_b[l * 768 + n];
            v8f acc;
#pragma unroll
            for (int r = 0; r < 8; ++r) acc[r] = bi;
#pragma unroll
            for (int kt = 0; kt < 8; ++kt) {
                v16bf a = load_A_lds(s_a + mt * 16 * CC + kt * 32, CC, lane);
                v16bf bm = load_Bt(Wq, CC, nt * 16, kt * 32, lane);
                acc = wmma_bf(a, bm, acc);
            }
            // region decided by the *scalar* nt -> uniform branch, no exec masking
            if (nt < 16) {                    // Q rows
#pragma unroll
                for (int r = 0; r < 8; ++r)
                    s_qkv[(mbase + r) * CC + n] = f2bf(acc[r]);
            } else if (nt < 32) {             // K rows
#pragma unroll
                for (int r = 0; r < 8; ++r)
                    s_qkv[NSAM * CC + (mbase + r) * CC + (n - 256)] = f2bf(acc[r]);
            } else {                          // V transposed [256][32]
#pragma unroll
                for (int r = 0; r < 8; ++r)
                    s_qkv[2 * NSAM * CC + (n - 512) * NSAM + (mbase + r)] = f2bf(acc[r]);
            }
        }
        __syncthreads();

        // ---- attention: wave = (head, m-tile) ----
        {
            int h  = wave >> 1, mt = wave & 1;
            const __bf16* sq  = s_qkv;
            const __bf16* sk  = s_qkv + NSAM * CC;
            const __bf16* svt = s_qkv + 2 * NSAM * CC;
            v8f sc0, sc1;
#pragma unroll
            for (int r = 0; r < 8; ++r) { sc0[r] = 0.f; sc1[r] = 0.f; }
#pragma unroll
            for (int kt = 0; kt < 2; ++kt) {
                v16bf a = load_A_lds(sq + mt * 16 * CC + h * DH + kt * 32, CC, lane);
                sc0 = wmma_bf(a, load_Bt(sk, CC, 0,  h * DH + kt * 32, lane), sc0);
                sc1 = wmma_bf(a, load_Bt(sk, CC, 16, h * DH + kt * 32, lane), sc1);
            }
#pragma unroll
            for (int r = 0; r < 8; ++r) {
                float v0 = sc0[r] * scale, v1 = sc1[r] * scale;
                float mx = fmaxf(v0, v1);
#pragma unroll
                for (int off = 1; off < 16; off <<= 1)
                    mx = fmaxf(mx, __shfl_xor(mx, off, 32));
                float e0 = __expf(v0 - mx), e1 = __expf(v1 - mx);
                float sm = e0 + e1;
#pragma unroll
                for (int off = 1; off < 16; off <<= 1)
                    sm += __shfl_xor(sm, off, 32);
                float inv = 1.0f / sm;
                int m = mt * 16 + ((lane >> 4) << 3) + r;
                s_p[h * NSAM * NSAM + m * NSAM + (lane & 15)]      = f2bf(e0 * inv);
                s_p[h * NSAM * NSAM + m * NSAM + 16 + (lane & 15)] = f2bf(e1 * inv);
            }
            // O = P @ V  (same wave wrote its P rows; LDS per-wave in-order)
            v16bf aP = load_A_lds(s_p + h * NSAM * NSAM + mt * 16 * NSAM, NSAM, lane);
#pragma unroll
            for (int dt = 0; dt < 4; ++dt) {
                v8f oc;
#pragma unroll
                for (int r = 0; r < 8; ++r) oc[r] = 0.f;
                oc = wmma_bf(aP, load_Bt(svt, NSAM, h * DH + dt * 16, 0, lane), oc);
#pragma unroll
                for (int r = 0; r < 8; ++r) {
                    int m = mt * 16 + ((lane >> 4) << 3) + r;
                    int n = h * DH + dt * 16 + (lane & 15);
                    s_a[m * CC + n] = f2bf(oc[r]);
                }
            }
        }
        __syncthreads();

        // ---- x = x_ln + O @ out_w^T + out_b ----
        gemm_rows<8, 16, true, GMODE_ADD_F32>(s_a, CC, out_wb + (size_t)l * CC * CC, CC,
                                              out_b + l * CC, s_x, CC, s_x, nullptr, CC,
                                              wave, lane);
        __syncthreads();

        layernorm_ws(s_x, s_a, ln2_g + l * CC, ln2_b + l * CC, tid);
        __syncthreads();

        // ---- FF1: relu(s_a @ ff1^T + b) -> bf16 hidden [32][512] in s_qkv ----
        gemm_rows<8, 32, false, GMODE_RELU_BF>(s_a, CC, ff1_wb + (size_t)l * 512 * CC, CC,
                                               ff1_b + l * 512, nullptr, 0,
                                               nullptr, s_qkv, 512, wave, lane);
        __syncthreads();

        // ---- FF2: x = x_ln2 + hidden @ ff2^T + b ----
        gemm_rows<16, 16, true, GMODE_ADD_F32>(s_qkv, 512, ff2_wb + (size_t)l * CC * 512, 512,
                                               ff2_b + l * CC, s_x, CC, s_x, nullptr, CC,
                                               wave, lane);
        __syncthreads();
    }

    // ---- max-pool over tokens + final FC ----
    {
        float mx = -1e30f;
        for (int t = 0; t < NSAM; ++t) mx = fmaxf(mx, s_x[t * CC + tid]);
        s_pool[tid] = mx;
    }
    __syncthreads();
    {
        float acc = fc_b[tid];
        const float* w = fc_w + tid * CC;
        for (int c = 0; c < CC; ++c) acc += w[c] * s_pool[c];
        out[((size_t)b * DOUT + tid) * NPT + g] = acc;
    }
}

// ---------- host launcher ----------
extern "C" void kernel_launch(void* const* d_in, const int* in_sizes, int n_in,
                              void* d_out, int out_size, void* d_ws, size_t ws_size,
                              hipStream_t stream) {
    (void)in_sizes; (void)n_in; (void)out_size; (void)ws_size;
    const float* xyz      = (const float*)d_in[0];
    const float* features = (const float*)d_in[1];
    const float* pe1_w = (const float*)d_in[2];
    const float* pe1_b = (const float*)d_in[3];
    const float* bn_g  = (const float*)d_in[4];
    const float* bn_b  = (const float*)d_in[5];
    const float* bn_m  = (const float*)d_in[6];
    const float* bn_v  = (const float*)d_in[7];
    const float* pe2_w = (const float*)d_in[8];
    const float* pe2_b = (const float*)d_in[9];
    const float* ln1_g = (const float*)d_in[10];
    const float* ln1_b = (const float*)d_in[11];
    const float* qkv_w = (const float*)d_in[12];
    const float* qkv_b = (const float*)d_in[13];
    const float* out_w = (const float*)d_in[14];
    const float* out_b = (const float*)d_in[15];
    const float* ln2_g = (const float*)d_in[16];
    const float* ln2_b = (const float*)d_in[17];
    const float* ff1_w = (const float*)d_in[18];
    const float* ff1_b = (const float*)d_in[19];
    const float* ff2_w = (const float*)d_in[20];
    const float* ff2_b = (const float*)d_in[21];
    const float* fc_w  = (const float*)d_in[22];
    const float* fc_b  = (const float*)d_in[23];

    char* ws = (char*)d_ws;
    size_t off = 0;
    auto take = [&](size_t bytes) -> void* {
        void* p = ws + off;
        off += (bytes + 255) & ~(size_t)255;
        return p;
    };
    int*    idx_buf  = (int*)take((size_t)BB * NPT * NSAM * sizeof(int));
    float*  nxyz_buf = (float*)take((size_t)BB * NPT * 3 * sizeof(float));
    __bf16* pe2_wb = (__bf16*)take((size_t)CC * 128 * 2);
    __bf16* qkv_wb = (__bf16*)take((size_t)LL * 768 * CC * 2);
    __bf16* out_wb = (__bf16*)take((size_t)LL * CC * CC * 2);
    __bf16* ff1_wb = (__bf16*)take((size_t)LL * 512 * CC * 2);
    __bf16* ff2_wb = (__bf16*)take((size_t)LL * CC * 512 * 2);

    float* o           = (float*)d_out;
    float* out_newxyz  = o;                                   // [B,NPT,3]
    float* out_feat    = o + (size_t)BB * NPT * 3;            // [B,DO,NPT]
    float* out_fpsidx  = out_feat + (size_t)BB * DOUT * NPT;  // [B,NPT] (as float)

    auto cvt = [&](const float* s, __bf16* d, int n) {
        cvt_bf16_kernel<<<(n + 255) / 256, 256, 0, stream>>>(s, d, n);
    };
    cvt(pe2_w, pe2_wb, CC * 128);
    cvt(qkv_w, qkv_wb, LL * 768 * CC);
    cvt(out_w, out_wb, LL * CC * CC);
    cvt(ff1_w, ff1_wb, LL * 512 * CC);
    cvt(ff2_w, ff2_wb, LL * CC * 512);

    fps_kernel<<<BB, 1024, 0, stream>>>(xyz, nxyz_buf, out_newxyz, out_fpsidx);
    ballq_kernel<<<(BB * NPT) / 8, 256, 0, stream>>>(xyz, nxyz_buf, idx_buf);
    group_kernel<<<BB * NPT, 256, 0, stream>>>(
        features, xyz, nxyz_buf, idx_buf,
        pe1_w, pe1_b, bn_g, bn_b, bn_m, bn_v, pe2_wb, pe2_b,
        ln1_g, ln1_b, qkv_wb, qkv_b, out_wb, out_b,
        ln2_g, ln2_b, ff1_wb, ff1_b, ff2_wb, ff2_b,
        fc_w, fc_b, out_feat);
}
